// CausalSelfAttention_13700945674540
// MI455X (gfx1250) — compile-verified, matrix operand order checked by reading, not run on hardware
//
#include <hip/hip_runtime.h>
#include <hip/hip_bf16.h>

// ---------------------------------------------------------------------------
// Causal self-attention (B=4, T=2048, C=1024, H=16, D=64) for gfx1250.
// bf16 WMMA (v_wmma_f32_16x16x32_bf16) with fp32 accumulation everywhere.
// GEMMs use async global->LDS staging (ASYNCcnt) for the shared B panel.
// ---------------------------------------------------------------------------

typedef float   f32x8   __attribute__((ext_vector_type(8)));
typedef __bf16  bf16x8  __attribute__((ext_vector_type(8)));
typedef __bf16  bf16x16 __attribute__((ext_vector_type(16)));

#define BB 4
#define TT 2048
#define CC 1024
#define HH 16
#define DD 64
#define MM (BB * TT)        // 8192
#define N3 (3 * CC)         // 3072

__device__ __forceinline__ f32x8 vzero8() {
    f32x8 z;
#pragma unroll
    for (int i = 0; i < 8; ++i) z[i] = 0.0f;
    return z;
}

// Assemble a 16-element bf16 A/B fragment from two contiguous 8-element chunks.
__device__ __forceinline__ bf16x16 load_frag(const __bf16* lo, const __bf16* hi) {
    const bf16x8 l = *reinterpret_cast<const bf16x8*>(lo);
    const bf16x8 h = *reinterpret_cast<const bf16x8*>(hi);
    bf16x16 out;
#pragma unroll
    for (int e = 0; e < 8; ++e) { out[e] = l[e]; out[e + 8] = h[e]; }
    return out;
}

__device__ __forceinline__ f32x8 wmma_bf16(bf16x16 a, bf16x16 b, f32x8 c) {
    return __builtin_amdgcn_wmma_f32_16x16x32_bf16(
        /*neg_a=*/false, a, /*neg_b=*/false, b,
        /*c_mod=*/(short)0, c, /*reuse_a=*/false, /*reuse_b=*/false);
}

// ---------------------------------------------------------------------------
// Prep: fp32 -> bf16 elementwise
// ---------------------------------------------------------------------------
__global__ void cvt_bf16_kernel(const float* __restrict__ src,
                                __bf16* __restrict__ dst, int n) {
    int i = blockIdx.x * blockDim.x + threadIdx.x;
    if (i < n) dst[i] = (__bf16)src[i];
}

// Prep: W[k][n] (KxN fp32) -> Wt[n][k] (NxK bf16)
__global__ void transpose_bf16_kernel(const float* __restrict__ W,
                                      __bf16* __restrict__ Wt, int K, int N) {
    int i = blockIdx.x * blockDim.x + threadIdx.x;
    if (i >= K * N) return;
    int k = i % K;
    int n = i / K;
    Wt[(size_t)n * K + k] = (__bf16)W[(size_t)k * N + n];
}

// ---------------------------------------------------------------------------
// GEMM: C[M][N] (fp32) = A[M][K] (bf16) * Bt[N][K]^T (bf16)
// 8 waves/block; each wave computes a 16x64 tile. grid = (N/64, M/128).
// Shared B panel (64 cols) is staged into LDS with async global->LDS copies,
// double-buffered so the next tile's DMA overlaps the current tile's WMMAs.
// ---------------------------------------------------------------------------
__global__ void gemm_bf16_kernel(const __bf16* __restrict__ A,
                                 const __bf16* __restrict__ Bt,
                                 float* __restrict__ C, int M, int N, int K) {
    __shared__ __bf16 Btile[2][64 * 32];   // 2 x 4KB tiles: [n_local][k_local]

    const int tid  = threadIdx.x;
    const int lane = tid & 31;
    const int wave = tid >> 5;
    const int r    = lane & 15;
    const int grp  = lane >> 4;            // 0: lanes 0-15, 1: lanes 16-31
    const int kOff = grp * 8;

    const int mBase = blockIdx.y * 128 + wave * 16;
    const int nBase = blockIdx.x * 64;

    f32x8 acc[4];
#pragma unroll
    for (int j = 0; j < 4; ++j) acc[j] = vzero8();

    const __bf16* aRow = A + (size_t)(mBase + r) * K;

    // async stage: 256 threads x 16B = 4KB = one 64x32 bf16 tile
    const int nLoc   = tid >> 2;           // 0..63 (row within B panel)
    const int kChunk = tid & 3;            // 0..3  (8 bf16 = 16B chunk)
    const __bf16* bSrcBase = Bt + (size_t)(nBase + nLoc) * K + kChunk * 8;

    auto stageB = [&](int buf, int k) {
        const __bf16* src = bSrcBase + k;
        unsigned lds_off =
            (unsigned)(uintptr_t)(&Btile[buf][nLoc * 32 + kChunk * 8]);
        asm volatile("global_load_async_to_lds_b128 %0, %1, off"
                     :: "v"(lds_off), "v"(src) : "memory");
    };

    stageB(0, 0);

    const int nK = K / 32;
    for (int ki = 0; ki < nK; ++ki) {
        asm volatile("s_wait_asynccnt 0x0" ::: "memory");
        __syncthreads();
        if (ki + 1 < nK) stageB((ki + 1) & 1, (ki + 1) * 32);

        const int k = ki * 32;
        bf16x16 af = load_frag(aRow + k + kOff, aRow + k + kOff + 16);
        const __bf16* bl = &Btile[ki & 1][0];
#pragma unroll
        for (int j = 0; j < 4; ++j) {
            const __bf16* bRow = bl + (j * 16 + r) * 32 + kOff;
            bf16x16 bf_ = load_frag(bRow, bRow + 16);
            acc[j] = wmma_bf16(af, bf_, acc[j]);
        }
        // no trailing barrier needed: next iteration's entry barrier orders
        // the buffer overwrite after all waves' reads of this iteration.
    }

    // D layout: element i of acc -> row (i + 8*grp), col r within tile
#pragma unroll
    for (int j = 0; j < 4; ++j) {
#pragma unroll
        for (int i = 0; i < 8; ++i) {
            C[(size_t)(mBase + i + 8 * grp) * N + nBase + j * 16 + r] = acc[j][i];
        }
    }
}

// ---------------------------------------------------------------------------
// RoPE + split: qkv fp32 [B*T][3C] -> q,k bf16 [B,H,T,D] (rotated), v^T bf16 [B,H,D,T]
// one thread per (b,t,h,d/2) pair
// ---------------------------------------------------------------------------
__global__ void rope_split_kernel(const float* __restrict__ qkv,
                                  __bf16* __restrict__ qb,
                                  __bf16* __restrict__ kb,
                                  __bf16* __restrict__ vt) {
    int idx = blockIdx.x * blockDim.x + threadIdx.x;   // B*T*H*32 = 2^22
    int d2 = idx & 31;
    int h  = (idx >> 5) & (HH - 1);
    int t  = (idx >> 9) & (TT - 1);
    int b  = idx >> 20;
    if (b >= BB) return;

    size_t row = (size_t)(b * TT + t) * N3;
    int c0 = h * DD + 2 * d2;

    float inv_freq = __powf(10000.0f, -((float)(2 * d2)) / (float)DD);
    float ang = (float)t * inv_freq;
    float sn, cs;
    __sincosf(ang, &sn, &cs);

    size_t qo = ((size_t)(b * HH + h) * TT + t) * DD + 2 * d2;

    float q1 = qkv[row + c0], q2 = qkv[row + c0 + 1];
    qb[qo]     = (__bf16)(q1 * cs - q2 * sn);
    qb[qo + 1] = (__bf16)(q1 * sn + q2 * cs);

    float k1 = qkv[row + CC + c0], k2 = qkv[row + CC + c0 + 1];
    kb[qo]     = (__bf16)(k1 * cs - k2 * sn);
    kb[qo + 1] = (__bf16)(k1 * sn + k2 * cs);

    float v1 = qkv[row + 2 * CC + c0], v2 = qkv[row + 2 * CC + c0 + 1];
    size_t vo = ((size_t)(b * HH + h) * DD + 2 * d2) * TT + t;
    vt[vo]      = (__bf16)v1;
    vt[vo + TT] = (__bf16)v2;
}

// ---------------------------------------------------------------------------
// Flash attention: one wave per 16-row q tile, 32 keys per iteration.
// q,k: [B,H,T,D] bf16 ; vt: [B,H,D,T] bf16 ; y: [B,T,H*D] bf16
// ---------------------------------------------------------------------------
__global__ void flash_attn_kernel(const __bf16* __restrict__ q,
                                  const __bf16* __restrict__ k,
                                  const __bf16* __restrict__ vt,
                                  __bf16* __restrict__ y) {
    __shared__ __bf16 ldsP[8 * 16 * 32];   // 8 waves x (16x32 P tile) = 8 KB

    const int lane = threadIdx.x & 31;
    const int wave = threadIdx.x >> 5;
    const int r    = lane & 15;
    const int grp  = lane >> 4;
    const int kOff = grp * 8;

    const int nQT = TT / 16;                       // 128 q tiles per head
    const int w   = blockIdx.x * 8 + wave;         // global wave-task id
    const int bh  = w / nQT;
    const int qt  = w % nQT;
    const int qBase = qt * 16;

    const __bf16* qhead = q  + (size_t)bh * TT * DD;
    const __bf16* khead = k  + (size_t)bh * TT * DD;
    const __bf16* vhead = vt + (size_t)bh * DD * TT;
    __bf16* ldsW = ldsP + wave * 512;

    // q fragments for this tile: rows qBase..qBase+15, d split 0..31 / 32..63
    const __bf16* qrow = qhead + (size_t)(qBase + r) * DD;
    const bf16x16 qa0 = load_frag(qrow + kOff,      qrow + kOff + 16);
    const bf16x16 qa1 = load_frag(qrow + 32 + kOff, qrow + 32 + kOff + 16);

    float mrun[8], lrun[8];
    f32x8 o[4];
#pragma unroll
    for (int i = 0; i < 8; ++i) { mrun[i] = -1e30f; lrun[i] = 0.0f; }
#pragma unroll
    for (int dt = 0; dt < 4; ++dt) o[dt] = vzero8();

    const float scale = 0.125f;   // 1/sqrt(64)

    for (int kbase = 0; kbase < qBase + 16; kbase += 32) {
        // ---- scores: two 16x16 tiles covering keys kbase..kbase+31 ----
        const __bf16* kr0 = khead + (size_t)(kbase + r) * DD;
        const __bf16* kr1 = khead + (size_t)(kbase + 16 + r) * DD;

        f32x8 s0 = vzero8(), s1 = vzero8();
        s0 = wmma_bf16(qa0, load_frag(kr0 + kOff,      kr0 + kOff + 16),      s0);
        s0 = wmma_bf16(qa1, load_frag(kr0 + 32 + kOff, kr0 + 32 + kOff + 16), s0);
        s1 = wmma_bf16(qa0, load_frag(kr1 + kOff,      kr1 + kOff + 16),      s1);
        s1 = wmma_bf16(qa1, load_frag(kr1 + 32 + kOff, kr1 + 32 + kOff + 16), s1);

        // ---- scale + causal mask + online softmax ----
#pragma unroll
        for (int i = 0; i < 8; ++i) {
            const int tq  = qBase + i + 8 * grp;     // this lane-element's query row
            float v0 = s0[i] * scale;
            float v1 = s1[i] * scale;
            v0 = (kbase + r      > tq) ? -1e30f : v0;
            v1 = (kbase + 16 + r > tq) ? -1e30f : v1;

            float mx = fmaxf(v0, v1);
#pragma unroll
            for (int msk = 1; msk < 16; msk <<= 1)
                mx = fmaxf(mx, __shfl_xor(mx, msk, 32));

            const float mnew  = fmaxf(mrun[i], mx);
            const float alpha = __expf(mrun[i] - mnew);
            mrun[i] = mnew;

            const float p0 = __expf(v0 - mnew);
            const float p1 = __expf(v1 - mnew);
            s0[i] = p0; s1[i] = p1;

            float rs = p0 + p1;
#pragma unroll
            for (int msk = 1; msk < 16; msk <<= 1)
                rs += __shfl_xor(rs, msk, 32);
            lrun[i] = lrun[i] * alpha + rs;

#pragma unroll
            for (int dt = 0; dt < 4; ++dt) o[dt][i] *= alpha;
        }

        // ---- transpose P (D-layout) to A-fragment layout via LDS ----
#pragma unroll
        for (int i = 0; i < 8; ++i) {
            ldsW[(i + 8 * grp) * 32 + r]      = (__bf16)s0[i];
            ldsW[(i + 8 * grp) * 32 + 16 + r] = (__bf16)s1[i];
        }
        asm volatile("s_wait_dscnt 0x0" ::: "memory");
        const __bf16* pr = ldsW + r * 32;
        const bf16x16 pf = load_frag(pr + kOff, pr + kOff + 16);

        // ---- O += P @ V ----
#pragma unroll
        for (int dt = 0; dt < 4; ++dt) {
            const __bf16* vrow = vhead + (size_t)(dt * 16 + r) * TT + kbase;
            const bf16x16 vf = load_frag(vrow + kOff, vrow + kOff + 16);
            o[dt] = wmma_bf16(pf, vf, o[dt]);
        }
    }

    // ---- finalize: y[b, t, h*64 + d] ----
    const int b = bh / HH;
    const int h = bh % HH;
#pragma unroll
    for (int i = 0; i < 8; ++i) {
        const float inv = 1.0f / lrun[i];
        const int t = qBase + i + 8 * grp;
        __bf16* yrow = y + ((size_t)(b * TT + t)) * CC + h * DD;
#pragma unroll
        for (int dt = 0; dt < 4; ++dt)
            yrow[dt * 16 + r] = (__bf16)(o[dt][i] * inv);
    }
}

// ---------------------------------------------------------------------------
// Host launcher
// ---------------------------------------------------------------------------
extern "C" void kernel_launch(void* const* d_in, const int* in_sizes, int n_in,
                              void* d_out, int out_size, void* d_ws, size_t ws_size,
                              hipStream_t stream) {
    const float* x  = (const float*)d_in[0];   // [B,T,C]
    const float* Wa = (const float*)d_in[1];   // [C, 3C]
    const float* Wp = (const float*)d_in[2];   // [C, C]
    float* out = (float*)d_out;                // [B,T,C] fp32

    char* ws = (char*)d_ws;
    size_t off = 0;
    auto take = [&](size_t bytes) -> void* {
        void* p = ws + off;
        off += (bytes + 255) & ~(size_t)255;
        return p;
    };

    __bf16* x_bf = (__bf16*)take((size_t)MM * CC * 2);          // 16 MB
    __bf16* Wa_t = (__bf16*)take((size_t)N3 * CC * 2);          //  6 MB
    __bf16* Wp_t = (__bf16*)take((size_t)CC * CC * 2);          //  2 MB
    float*  qkv  = (float*) take((size_t)MM * N3 * 4);          // 96 MB
    __bf16* qb   = (__bf16*)take((size_t)BB * HH * TT * DD * 2);// 16 MB
    __bf16* kb   = (__bf16*)take((size_t)BB * HH * TT * DD * 2);// 16 MB
    __bf16* vt   = (__bf16*)take((size_t)BB * HH * DD * TT * 2);// 16 MB
    __bf16* yb   = (__bf16*)take((size_t)MM * CC * 2);          // 16 MB

    // 1) convert x to bf16
    {
        int n = MM * CC;
        cvt_bf16_kernel<<<(n + 255) / 256, 256, 0, stream>>>(x, x_bf, n);
    }
    // 2) transpose weights to [N][K] bf16
    {
        int n = CC * N3;
        transpose_bf16_kernel<<<(n + 255) / 256, 256, 0, stream>>>(Wa, Wa_t, CC, N3);
    }
    {
        int n = CC * CC;
        transpose_bf16_kernel<<<(n + 255) / 256, 256, 0, stream>>>(Wp, Wp_t, CC, CC);
    }
    // 3) QKV projection: qkv = x @ W_attn
    gemm_bf16_kernel<<<dim3(N3 / 64, MM / 128), 256, 0, stream>>>(
        x_bf, Wa_t, qkv, MM, N3, CC);
    // 4) RoPE + head split/transposes
    {
        int n = BB * TT * HH * 32;
        rope_split_kernel<<<(n + 255) / 256, 256, 0, stream>>>(qkv, qb, kb, vt);
    }
    // 5) causal flash attention
    {
        int waves = BB * HH * (TT / 16);   // 8192
        flash_attn_kernel<<<waves / 8, 256, 0, stream>>>(qb, kb, vt, yb);
    }
    // 6) output projection: out = y @ W_proj
    gemm_bf16_kernel<<<dim3(CC / 64, MM / 128), 256, 0, stream>>>(
        yb, Wp_t, out, MM, CC, CC);
}
